// T5Attention_86543591015038
// MI455X (gfx1250) — matrix operand order repaired and checked by reading, not compiled, not run on hardware
//
#include <hip/hip_runtime.h>

typedef __attribute__((ext_vector_type(16))) __bf16 bf16x16;
typedef __attribute__((ext_vector_type(8)))  float  v8f;

#define WMMA_BF16(a, b, c) \
  __builtin_amdgcn_wmma_f32_16x16x32_bf16(false, (a), false, (b), (short)0, (c), false, false)

// ---------- helpers ----------

__device__ __forceinline__ unsigned short f2bf(float f) {
  unsigned int u = __float_as_uint(f);
  u += 0x7FFFu + ((u >> 16) & 1u);      // round to nearest even
  return (unsigned short)(u >> 16);
}

union FragU { bf16x16 v; uint4 u[2]; };

// A-matrix 16x32 bf16 fragment: lane holds row (lane&15); K chunks at
// half*8 .. half*8+7 and 16+half*8 .. (two 16B loads). `p` = start of 32-K row.
__device__ __forceinline__ bf16x16 ld_a_frag(const unsigned short* p, int halfk) {
  FragU f;
  f.u[0] = *(const uint4*)(p + halfk * 8);
  f.u[1] = *(const uint4*)(p + 16 + halfk * 8);
  return f.v;
}

// B-matrix 32x16 bf16 fragment: lane holds col (lane&15); K chunks at
// half*8 .. half*8+7 and 16+half*8 .. (two 16B loads). `p` = start of the 32-K run for this column (row of B^T).
__device__ __forceinline__ bf16x16 ld_b_frag(const unsigned short* p, int halfk) {
  FragU f;
  f.u[0] = *(const uint4*)(p + halfk * 8);
  f.u[1] = *(const uint4*)(p + 16 + halfk * 8);
  return f.v;
}

// ---------- kernel 1: relative-position bias LUT ----------
// biasTab[h][d + 2047] for d = key - query in [-2047, 2047]
__global__ __launch_bounds__(256) void bias_precompute(
    const float* __restrict__ rel_emb, float* __restrict__ biasTab) {
  int i = blockIdx.x * blockDim.x + threadIdx.x;
  if (i >= 16 * 4095) return;
  int h   = i / 4095;
  int rel = (i % 4095) - 2047;          // memory - context
  int n   = -rel;
  int ret = 0;
  if (n < 0) { ret = 16; n = -n; }
  int v;
  if (n < 8) {
    v = n;
  } else {
    // 8 + log(n/8)/log(16) * 8, clamped to 15
    v = 8 + (int)(__logf((float)n * 0.125f) * (8.0f / 2.7725887222397812f));
    if (v > 15) v = 15;
  }
  biasTab[h * 4095 + (rel + 2047)] = rel_emb[(ret + v) * 16 + h];
}

// ---------- kernel 2: QKV projections (templated on output layout) ----------
// X[8192,1024] fp32 * W[1024,1024] fp32 -> bf16
//   MID=0/1: Q/K as [b,h,s,d];  MID=2: V transposed as [b,h,d,s]
template <int MID>
__global__ __launch_bounds__(256) void qkv_gemm(
    const float* __restrict__ X, const float* __restrict__ W,
    unsigned short* __restrict__ Dst) {
  __shared__ __align__(16) unsigned short As[128][40];  // [m][k]
  __shared__ __align__(16) unsigned short Bs[128][40];  // B^T: [n][k]

  const int tid = threadIdx.x;
  const int lane = tid & 31, wave = tid >> 5;
  const int wm = wave >> 2, wn = wave & 3;            // 2 x 4 wave grid
  const int m0 = blockIdx.x * 128, n0 = blockIdx.y * 128;
  const int row = lane & 15, halfk = lane >> 4;

  v8f acc[4][2] = {};

  for (int kt = 0; kt < 1024; kt += 32) {
    // stage A tile (fp32 -> bf16)
#pragma unroll
    for (int i = 0; i < 4; ++i) {
      int idx = tid + i * 256;
      int r = idx >> 3, c = idx & 7;
      float4 f = *(const float4*)(X + (size_t)(m0 + r) * 1024 + kt + c * 4);
      unsigned short* d = &As[r][c * 4];
      d[0] = f2bf(f.x); d[1] = f2bf(f.y); d[2] = f2bf(f.z); d[3] = f2bf(f.w);
    }
    // stage B tile transposed (fp32 -> bf16)
#pragma unroll
    for (int i = 0; i < 4; ++i) {
      int idx = tid + i * 256;
      int r = idx >> 5, c = idx & 31;
      float4 f = *(const float4*)(W + (size_t)(kt + r) * 1024 + n0 + c * 4);
      Bs[c * 4 + 0][r] = f2bf(f.x); Bs[c * 4 + 1][r] = f2bf(f.y);
      Bs[c * 4 + 2][r] = f2bf(f.z); Bs[c * 4 + 3][r] = f2bf(f.w);
    }
    __syncthreads();

    bf16x16 af[4], bfr[2];
#pragma unroll
    for (int mi = 0; mi < 4; ++mi)
      af[mi] = ld_a_frag(&As[wm * 64 + mi * 16 + row][0], halfk);
#pragma unroll
    for (int ni = 0; ni < 2; ++ni)
      bfr[ni] = ld_b_frag(&Bs[wn * 32 + ni * 16 + row][0], halfk);
#pragma unroll
    for (int mi = 0; mi < 4; ++mi)
#pragma unroll
      for (int ni = 0; ni < 2; ++ni)
        acc[mi][ni] = WMMA_BF16(af[mi], bfr[ni], acc[mi][ni]);
    __syncthreads();
  }

  // epilogue (branch-free; layout chosen at compile time)
#pragma unroll
  for (int mi = 0; mi < 4; ++mi) {
#pragma unroll
    for (int ni = 0; ni < 2; ++ni) {
      int n = n0 + wn * 32 + ni * 16 + row;     // inner index 0..1023
      int h = n >> 6, dd = n & 63;
      int r_base = m0 + wm * 64 + mi * 16 + halfk * 8;   // 8 consecutive tokens
      int b = r_base >> 11, s = r_base & 2047;
      if (MID == 2) {
        // Vt[b,h,dd, s..s+7]: pack 8 bf16 -> one 16B store
        union { unsigned short s16[8]; uint4 u; } pk;
#pragma unroll
        for (int e = 0; e < 8; ++e) pk.s16[e] = f2bf(acc[mi][ni][e]);
        *(uint4*)(Dst + ((size_t)(b * 16 + h) * 64 + dd) * 2048 + s) = pk.u;
      } else {
        // Q/K[b,h,s..s+7,dd]: scattered bf16, stride 64
        unsigned short* base =
            Dst + ((size_t)(b * 16 + h) * 2048 + s) * 64 + dd;
#pragma unroll
        for (int e = 0; e < 8; ++e) base[(size_t)e * 64] = f2bf(acc[mi][ni][e]);
      }
    }
  }
}

// ---------- kernel 3: flash attention with T5 bias ----------
__global__ __launch_bounds__(256) void flash_attn(
    const unsigned short* __restrict__ Q, const unsigned short* __restrict__ K,
    const unsigned short* __restrict__ Vt, const float* __restrict__ biasTab,
    unsigned short* __restrict__ Ctx) {
  __shared__ __align__(16) unsigned short Ks[32][72];      // [key][d]
  __shared__ __align__(16) unsigned short Vs[64][40];      // [d][key]
  __shared__ __align__(16) unsigned short Ps[8][16][40];   // per-wave P tile

  const int b = blockIdx.z, h = blockIdx.y, q0 = blockIdx.x * 128;
  const int tid = threadIdx.x, lane = tid & 31, wave = tid >> 5;
  const int row = lane & 15, halfk = lane >> 4;

  const unsigned short* Qh = Q + (size_t)(b * 16 + h) * 2048 * 64;
  const unsigned short* Kh = K + (size_t)(b * 16 + h) * 2048 * 64;
  const unsigned short* Vh = Vt + (size_t)(b * 16 + h) * 64 * 2048;
  const float* biasH = biasTab + h * 4095 + 2047;

  // each wave owns 16 query rows; Q fragment is loop-invariant
  const unsigned short* qrow = Qh + (size_t)(q0 + wave * 16 + row) * 64;
  bf16x16 qf0 = ld_a_frag(qrow, halfk);         // d 0..31
  bf16x16 qf1 = ld_a_frag(qrow + 32, halfk);    // d 32..63

  v8f o[4] = {};
  float mrow[8], lrow[8];
#pragma unroll
  for (int e = 0; e < 8; ++e) { mrow[e] = -3.0e38f; lrow[e] = 0.0f; }

  const int kr = tid >> 3, kc = tid & 7;  // K panel: 32 rows x 8 uint4
  const int vr = tid >> 2, vc = tid & 3;  // V panel: 64 rows x 4 uint4

  for (int kb = 0; kb < 2048; kb += 32) {
    *(uint4*)(&Ks[kr][kc * 8]) = *(const uint4*)(Kh + (size_t)(kb + kr) * 64 + kc * 8);
    *(uint4*)(&Vs[vr][vc * 8]) = *(const uint4*)(Vh + (size_t)vr * 2048 + kb + vc * 8);
    if (kb + 32 < 2048) {
      __builtin_prefetch(Kh + (size_t)(kb + 32 + kr) * 64 + kc * 8, 0, 1);
      __builtin_prefetch(Vh + (size_t)vr * 2048 + kb + 32 + vc * 8, 0, 1);
    }
    __syncthreads();

    // scores: two 16x16 tiles over 32 keys (K-dim = 64 -> 2 WMMAs each)
    v8f s0 = {}, s1 = {};
    {
      bf16x16 b00 = ld_b_frag(&Ks[row][0], halfk);
      bf16x16 b01 = ld_b_frag(&Ks[row][32], halfk);
      bf16x16 b10 = ld_b_frag(&Ks[16 + row][0], halfk);
      bf16x16 b11 = ld_b_frag(&Ks[16 + row][32], halfk);
      s0 = WMMA_BF16(qf0, b00, s0);
      s0 = WMMA_BF16(qf1, b01, s0);
      s1 = WMMA_BF16(qf0, b10, s1);
      s1 = WMMA_BF16(qf1, b11, s1);
    }

    // T5 relative-position bias via Toeplitz LUT
    const int key0 = kb + row, key1 = key0 + 16;
    const int qbase = q0 + wave * 16 + halfk * 8;
    float p0[8], p1[8], rm[8];
#pragma unroll
    for (int e = 0; e < 8; ++e) {
      int qe = qbase + e;
      float v0 = s0[e] + biasH[key0 - qe];
      float v1 = s1[e] + biasH[key1 - qe];
      p0[e] = v0; p1[e] = v1;
      rm[e] = fmaxf(v0, v1);
    }
    // row max over 16 lanes of the half (xor<=8 stays within the half)
#pragma unroll
    for (int e = 0; e < 8; ++e) {
#pragma unroll
      for (int off = 1; off < 16; off <<= 1)
        rm[e] = fmaxf(rm[e], __shfl_xor(rm[e], off, 32));
    }
    // online softmax update
#pragma unroll
    for (int e = 0; e < 8; ++e) {
      float mn = fmaxf(mrow[e], rm[e]);
      float sc = __expf(mrow[e] - mn);
      mrow[e] = mn;
      lrow[e] *= sc;
#pragma unroll
      for (int t = 0; t < 4; ++t) o[t][e] *= sc;
      float e0 = __expf(p0[e] - mn);
      float e1 = __expf(p1[e] - mn);
      p0[e] = e0; p1[e] = e1;
      float rs = e0 + e1;
#pragma unroll
      for (int off = 1; off < 16; off <<= 1)
        rs += __shfl_xor(rs, off, 32);
      lrow[e] += rs;
    }

    // C-layout -> A-layout for P via per-wave LDS region
#pragma unroll
    for (int e = 0; e < 8; ++e) {
      Ps[wave][e + halfk * 8][row]      = f2bf(p0[e]);
      Ps[wave][e + halfk * 8][16 + row] = f2bf(p1[e]);
    }
    asm volatile("s_wait_dscnt 0" ::: "memory");   // in-wave cross-lane LDS hazard

    bf16x16 pf = ld_a_frag(&Ps[wave][row][0], halfk);
#pragma unroll
    for (int t = 0; t < 4; ++t) {
      bf16x16 vf = ld_b_frag(&Vs[t * 16 + row][0], halfk);
      o[t] = WMMA_BF16(pf, vf, o[t]);
    }
    __syncthreads();
  }

  // normalize and emit context as [b, s, h*64 + d] bf16
#pragma unroll
  for (int t = 0; t < 4; ++t) {
#pragma unroll
    for (int e = 0; e < 8; ++e) {
      int qa = q0 + wave * 16 + e + halfk * 8;
      float val = o[t][e] / lrow[e];
      Ctx[((size_t)b * 2048 + qa) * 1024 + h * 64 + t * 16 + row] = f2bf(val);
    }
  }
}

// ---------- kernel 4: output projection ----------
// Ctx[8192,1024] bf16 * Wo[1024,1024] fp32 -> Out fp32
__global__ __launch_bounds__(256) void out_gemm(
    const unsigned short* __restrict__ Ctx, const float* __restrict__ Wo,
    float* __restrict__ Out) {
  __shared__ __align__(16) unsigned short As[128][40];
  __shared__ __align__(16) unsigned short Bs[128][40];

  const int tid = threadIdx.x;
  const int lane = tid & 31, wave = tid >> 5;
  const int wm = wave >> 2, wn = wave & 3;
  const int m0 = blockIdx.x * 128, n0 = blockIdx.y * 128;
  const int row = lane & 15, halfk = lane >> 4;

  v8f acc[4][2] = {};

  for (int kt = 0; kt < 1024; kt += 32) {
#pragma unroll
    for (int i = 0; i < 2; ++i) {
      int idx = tid + i * 256;
      int r = idx >> 2, c = idx & 3;
      *(uint4*)(&As[r][c * 8]) =
          *(const uint4*)(Ctx + (size_t)(m0 + r) * 1024 + kt + c * 8);
    }
#pragma unroll
    for (int i = 0; i < 4; ++i) {
      int idx = tid + i * 256;
      int r = idx >> 5, c = idx & 31;
      float4 f = *(const float4*)(Wo + (size_t)(kt + r) * 1024 + n0 + c * 4);
      Bs[c * 4 + 0][r] = f2bf(f.x); Bs[c * 4 + 1][r] = f2bf(f.y);
      Bs[c * 4 + 2][r] = f2bf(f.z); Bs[c * 4 + 3][r] = f2bf(f.w);
    }
    __syncthreads();

    bf16x16 af[4], bfr[2];
#pragma unroll
    for (int mi = 0; mi < 4; ++mi)
      af[mi] = ld_a_frag(&As[wm * 64 + mi * 16 + row][0], halfk);
#pragma unroll
    for (int ni = 0; ni < 2; ++ni)
      bfr[ni] = ld_b_frag(&Bs[wn * 32 + ni * 16 + row][0], halfk);
#pragma unroll
    for (int mi = 0; mi < 4; ++mi)
#pragma unroll
      for (int ni = 0; ni < 2; ++ni)
        acc[mi][ni] = WMMA_BF16(af[mi], bfr[ni], acc[mi][ni]);
    __syncthreads();
  }

#pragma unroll
  for (int mi = 0; mi < 4; ++mi) {
#pragma unroll
    for (int ni = 0; ni < 2; ++ni) {
      int n = n0 + wn * 32 + ni * 16 + row;
#pragma unroll
      for (int e = 0; e < 8; ++e) {
        int r = m0 + wm * 64 + mi * 16 + e + halfk * 8;
        Out[(size_t)r * 1024 + n] = acc[mi][ni][e];
      }
    }
  }
}

// ---------- launch ----------
extern "C" void kernel_launch(void* const* d_in, const int* in_sizes, int n_in,
                              void* d_out, int out_size, void* d_ws, size_t ws_size,
                              hipStream_t stream) {
  (void)in_sizes; (void)n_in; (void)out_size; (void)ws_size;
  const float* X   = (const float*)d_in[0];  // [4,2048,1024]
  const float* Wq  = (const float*)d_in[1];
  const float* Wk  = (const float*)d_in[2];
  const float* Wv  = (const float*)d_in[3];
  const float* Wo  = (const float*)d_in[4];
  const float* rel = (const float*)d_in[5];  // [32,16]

  const size_t HEAD_ELEMS = (size_t)4 * 16 * 2048 * 64;  // 8,388,608
  unsigned short* Q   = (unsigned short*)d_ws;
  unsigned short* K   = Q + HEAD_ELEMS;
  unsigned short* Vt  = K + HEAD_ELEMS;
  unsigned short* Ctx = Vt + HEAD_ELEMS;
  float* biasTab = (float*)(Ctx + HEAD_ELEMS);           // 16*4095 floats

  bias_precompute<<<(16 * 4095 + 255) / 256, 256, 0, stream>>>(rel, biasTab);
  qkv_gemm<0><<<dim3(64, 8), 256, 0, stream>>>(X, Wq, Q);
  qkv_gemm<1><<<dim3(64, 8), 256, 0, stream>>>(X, Wk, K);
  qkv_gemm<2><<<dim3(64, 8), 256, 0, stream>>>(X, Wv, Vt);
  flash_attn<<<dim3(16, 16, 4), 256, 0, stream>>>(Q, K, Vt, biasTab, Ctx);
  out_gemm<<<dim3(64, 8), 256, 0, stream>>>(Ctx, Wo, (float*)d_out);
}
